// QueueNet_64493228916790
// MI455X (gfx1250) — matrix operand order repaired and checked by reading, not compiled, not run on hardware
//
#include <hip/hip_runtime.h>
#include <hip/hip_bf16.h>

typedef __attribute__((ext_vector_type(16))) __bf16          v16bf;
typedef __attribute__((ext_vector_type(8)))  float           v8f;
typedef __attribute__((ext_vector_type(4)))  float           f4;

#define NTILE    4096      // 65536 positions / 16
#define NFRAG    280
#define FRAG_WS  0         // 16 frags: start_w^T  (kt0..7, nt0..1)
#define FRAG_WG  16        // 40 frags: gate_w A-layout, per layer
#define FRAG_WRT 56        // 80 frags: res_w^T (layer, nt0..1)
#define FRAG_WC1 136       // 16 frags: end_conv1_w^T (kt0..1, nt0..7)
#define FRAG_WC2 152       // 128 frags: end_conv2_w^T (kt0..7, nt0..15)

union BF { __bf16 e[16]; v16bf b; };

__device__ __forceinline__ __bf16 f2bf(float f) { return (__bf16)f; }  // native cvt, RNE

// mish(x) = x*tanh(softplus(x)) = x * n/(n+2), n = e^x*(e^x+2); clamp keeps ratio->1
__device__ __forceinline__ float mishf(float x) {
  float e = __expf(fminf(x, 15.f));
  float n = e * (e + 2.f);
  return x * n * __builtin_amdgcn_rcpf(n + 2.f);
}

__device__ __forceinline__ v8f wmma_bf16(v16bf a, v16bf b, v8f c) {
  return __builtin_amdgcn_wmma_f32_16x16x32_bf16(false, a, false, b, (short)0, c, false, false);
}

__device__ __forceinline__ BF load_frag(const __bf16* wf, int f, int lane) {
  BF r; r.b = *(const v16bf*)(wf + (size_t)f * 512 + lane * 16); return r;
}

// A-matrix (16-bit, 16x32) per-lane K map from ISA 7.12.2
__device__ __forceinline__ int kmapA(int h, int j) { return (j < 8) ? (8*h + j) : (8*h + j + 8); }
// B-matrix (16-bit, 32x16) per-lane K map: same kmapA order as A

//==================== weight fragment pre-pack ====================
__global__ void prep_frags_k(const float* __restrict__ sw, const float* __restrict__ gw,
                             const float* __restrict__ rw, const float* __restrict__ c1w,
                             const float* __restrict__ c2w, __bf16* __restrict__ wf) {
  int gid = blockIdx.x * blockDim.x + threadIdx.x;
  int f = gid >> 5, lane = gid & 31;
  if (f >= NFRAG) return;
  int h = lane >> 4, n = lane & 15;
  __bf16 o[16];
  #pragma unroll
  for (int j = 0; j < 16; ++j) {
    float v;
    if (f < FRAG_WG) {                        // Ws^T  B-frags
      int kt = f >> 1, nt = f & 1;
      v = sw[(16*nt + n) * 256 + 32*kt + kmapA(h, j)];
    } else if (f < FRAG_WRT) {                // Wg A-frags
      int i = f - FRAG_WG;
      v = gw[i * 512 + n * 32 + kmapA(h, j)];
    } else if (f < FRAG_WC1) {                // Wr^T B-frags
      int q = f - FRAG_WRT, i = q >> 1, nt = q & 1;
      v = rw[i * 1024 + (16*nt + n) * 32 + kmapA(h, j)];
    } else if (f < FRAG_WC2) {                // Wc1^T B-frags
      int q = f - FRAG_WC1, kt = q >> 3, nt = q & 7;
      v = c1w[(16*nt + n) * 64 + 32*kt + kmapA(h, j)];
    } else {                                  // Wc2^T B-frags
      int q = f - FRAG_WC2, kt = q >> 4, nt = q & 15;
      v = c2w[(16*nt + n) * 256 + 32*kt + kmapA(h, j)];
    }
    o[j] = f2bf(v);
  }
  __bf16* p = wf + (size_t)f * 512 + lane * 16;
  #pragma unroll
  for (int j = 0; j < 16; ++j) p[j] = o[j];
}

//==================== start conv: r = Ws @ x + b ====================
// r stored tiled: r[tile*512 + c*16 + pos], c in [0,32), pos in [0,16)
__global__ void start_k(const float* __restrict__ x, const __bf16* __restrict__ wf,
                        const float* __restrict__ sb, float* __restrict__ r) {
  int lane = threadIdx.x & 31;
  int wave = (blockIdx.x * blockDim.x + threadIdx.x) >> 5;
  int nw = (gridDim.x * blockDim.x) >> 5;
  int h = lane >> 4, nn = lane & 15;
  float b0 = sb[nn], b1 = sb[16 + nn];
  for (int tile = wave; tile < NTILE; tile += nw) {
    int n0 = tile * 16, bidx = n0 >> 13, l0 = n0 & 8191;
    const float* xp = x + (size_t)bidx * 2097152 + l0 + nn;  // lane row = position nn
    v8f D0 = {0,0,0,0,0,0,0,0};
    v8f D1 = {0,0,0,0,0,0,0,0};
    for (int kt = 0; kt < 8; ++kt) {          // K-outer: only one A frag live
      BF A;
      #pragma unroll
      for (int j = 0; j < 16; ++j) {
        int ch = 32 * kt + kmapA(h, j);
        A.e[j] = f2bf(xp[(size_t)ch * 8192]);
      }
      D0 = wmma_bf16(A.b, load_frag(wf, FRAG_WS + kt*2 + 0, lane).b, D0);
      D1 = wmma_bf16(A.b, load_frag(wf, FRAG_WS + kt*2 + 1, lane).b, D1);
    }
    {
      float* rp = r + (size_t)tile * 512 + nn * 16 + 8*h;
      f4 o0 = {D0[0]+b0, D0[1]+b0, D0[2]+b0, D0[3]+b0};
      f4 o1 = {D0[4]+b0, D0[5]+b0, D0[6]+b0, D0[7]+b0};
      *(f4*)rp = o0; *(f4*)(rp + 4) = o1;
    }
    {
      float* rp = r + (size_t)tile * 512 + (16 + nn) * 16 + 8*h;
      f4 o0 = {D1[0]+b1, D1[1]+b1, D1[2]+b1, D1[3]+b1};
      f4 o1 = {D1[4]+b1, D1[5]+b1, D1[6]+b1, D1[7]+b1};
      *(f4*)rp = o0; *(f4*)(rp + 4) = o1;
    }
  }
}

//==================== per-layer: t = Wr @ concat_mish(Wg @ r + bg) + br, stats ====================
__global__ void layer_k(const float* __restrict__ r, float* __restrict__ t,
                        const __bf16* __restrict__ wf,
                        const float* __restrict__ gbias, const float* __restrict__ rbias,
                        int li, float* __restrict__ pbuf) {
  __shared__ float sh[1024];
  int lane = threadIdx.x & 31, wid = threadIdx.x >> 5;
  int wave = (blockIdx.x * blockDim.x + threadIdx.x) >> 5;
  int nw = (gridDim.x * blockDim.x) >> 5;
  int h = lane >> 4, nn = lane & 15;
  BF Ag  = load_frag(wf, FRAG_WG + li, lane);
  BF Bw0 = load_frag(wf, FRAG_WRT + li*2,     lane);
  BF Bw1 = load_frag(wf, FRAG_WRT + li*2 + 1, lane);
  float gb[8];
  #pragma unroll
  for (int v = 0; v < 8; ++v) gb[v] = gbias[li*16 + 8*h + v];
  float rb0 = rbias[li*32 + nn], rb1 = rbias[li*32 + 16 + nn];
  float s0 = 0, q0 = 0, s1 = 0, q1 = 0;
  for (int tile = wave; tile < NTILE; tile += nw) {
    const float* rt = r + (size_t)tile * 512;
    if (tile + nw < NTILE) __builtin_prefetch(r + (size_t)(tile + nw) * 512 + lane * 16, 0, 1);
    BF Br;                                            // B = r-tile (32 x 16 pos)
    #pragma unroll
    for (int j = 0; j < 16; ++j) Br.e[j] = f2bf(rt[kmapA(h, j) * 16 + nn]);
    v8f Dg = {0,0,0,0,0,0,0,0};
    Dg = wmma_bf16(Ag.b, Br.b, Dg);                   // g = Wg @ r  (lane = pos)
    BF A2;                                            // in-lane concat_mish -> h^T A-frag
    #pragma unroll
    for (int v = 0; v < 8; ++v) {
      float g = Dg[v] + gb[v];
      A2.e[v]     = f2bf(mishf(g));
      A2.e[v + 8] = f2bf(mishf(-g));
    }
    #pragma unroll
    for (int nt = 0; nt < 2; ++nt) {
      v8f D = {0,0,0,0,0,0,0,0};
      D = wmma_bf16(A2.b, (nt ? Bw1.b : Bw0.b), D);   // t^T = h^T @ Wr^T
      float bias = nt ? rb1 : rb0;
      float* tp = t + (size_t)tile * 512 + (16*nt + nn) * 16 + 8*h;
      float vs[8];
      #pragma unroll
      for (int v = 0; v < 8; ++v) vs[v] = D[v] + bias;
      f4 o0 = {vs[0],vs[1],vs[2],vs[3]}, o1 = {vs[4],vs[5],vs[6],vs[7]};
      *(f4*)tp = o0; *(f4*)(tp + 4) = o1;
      float ls = 0, lq = 0;
      #pragma unroll
      for (int v = 0; v < 8; ++v) { ls += vs[v]; lq += vs[v]*vs[v]; }
      if (nt == 0) { s0 += ls; q0 += lq; } else { s1 += ls; q1 += lq; }
    }
  }
  sh[wid*128 + lane*4 + 0] = s0; sh[wid*128 + lane*4 + 1] = q0;
  sh[wid*128 + lane*4 + 2] = s1; sh[wid*128 + lane*4 + 3] = q1;
  __syncthreads();
  if (threadIdx.x < 64) {                             // deterministic fixed-order reduce
    int c = threadIdx.x >> 1, k = threadIdx.x & 1;
    int nt = c >> 4, cl = c & 15;
    float acc = 0;
    for (int w = 0; w < 8; ++w)
      for (int p = 0; p < 2; ++p)
        acc += sh[w*128 + (cl + 16*p)*4 + nt*2 + k];
    pbuf[blockIdx.x * 256 + c*2 + k] = acc;
  }
}

//==================== BN finalize: scale/shift from partial sums ====================
__global__ void finalize_bn_k(const float* __restrict__ pbuf, int nblk, int C,
                              const float* __restrict__ g, const float* __restrict__ b,
                              float* __restrict__ scale, float* __restrict__ shift) {
  int c = threadIdx.x;
  if (c >= C) return;
  float s = 0, q = 0;
  for (int i = 0; i < nblk; ++i) { s += pbuf[i*256 + c*2]; q += pbuf[i*256 + c*2 + 1]; }
  float m  = s * (1.0f / 65536.0f);
  float vr = q * (1.0f / 65536.0f) - m*m;
  float is = rsqrtf(vr + 1e-5f);
  float sc = g[c] * is;
  scale[c] = sc;
  shift[c] = b[c] - m * sc;
}

//==================== residual update: r += t*scale + shift ====================
__global__ void bnres_k(float* __restrict__ r, const float* __restrict__ t,
                        const float* __restrict__ scale, const float* __restrict__ shift) {
  int i = blockIdx.x * blockDim.x + threadIdx.x;
  int stride = gridDim.x * blockDim.x;
  for (int idx4 = i; idx4 < 2097152 / 4; idx4 += stride) {
    int c = (idx4 >> 2) & 31;
    float sc = scale[c], sf = shift[c];
    f4 tv = ((const f4*)t)[idx4];
    f4 rv = ((f4*)r)[idx4];
    rv.x += tv.x*sc + sf; rv.y += tv.y*sc + sf;
    rv.z += tv.z*sc + sf; rv.w += tv.w*sc + sf;
    ((f4*)r)[idx4] = rv;
  }
}

//==================== stats of r (for end_bn1) ====================
__global__ void stats_r_k(const float* __restrict__ r, float* __restrict__ pbuf) {
  __shared__ float sh[1024];
  int lane = threadIdx.x & 31, wid = threadIdx.x >> 5;
  int wave = (blockIdx.x * blockDim.x + threadIdx.x) >> 5;
  int nw = (gridDim.x * blockDim.x) >> 5;
  int h = lane >> 4, nn = lane & 15;
  float s0 = 0, q0 = 0, s1 = 0, q1 = 0;
  for (int tile = wave; tile < NTILE; tile += nw) {
    const float* rt = r + (size_t)tile * 512;
    #pragma unroll
    for (int v = 0; v < 8; ++v) {
      float a = rt[nn*16 + 8*h + v];
      float b = rt[(16 + nn)*16 + 8*h + v];
      s0 += a; q0 += a*a; s1 += b; q1 += b*b;
    }
  }
  sh[wid*128 + lane*4 + 0] = s0; sh[wid*128 + lane*4 + 1] = q0;
  sh[wid*128 + lane*4 + 2] = s1; sh[wid*128 + lane*4 + 3] = q1;
  __syncthreads();
  if (threadIdx.x < 64) {
    int c = threadIdx.x >> 1, k = threadIdx.x & 1;
    int nt = c >> 4, cl = c & 15;
    float acc = 0;
    for (int w = 0; w < 8; ++w)
      for (int p = 0; p < 2; ++p)
        acc += sh[w*128 + (cl + 16*p)*4 + nt*2 + k];
    pbuf[blockIdx.x * 256 + c*2 + k] = acc;
  }
}

//==================== end stage 1: u = Wc1 @ concat_mish(bn1(r)) + b, stats of u ====================
__global__ void end1_k(const float* __restrict__ r, float* __restrict__ u,
                       const __bf16* __restrict__ wf,
                       const float* __restrict__ c1b,
                       const float* __restrict__ scale1, const float* __restrict__ shift1,
                       float* __restrict__ pbuf) {
  __shared__ float sh[4096];
  int lane = threadIdx.x & 31, wid = threadIdx.x >> 5;
  int wave = (blockIdx.x * blockDim.x + threadIdx.x) >> 5;
  int nw = (gridDim.x * blockDim.x) >> 5;
  int h = lane >> 4, nn = lane & 15;
  float sc[16], sf[16];
  #pragma unroll
  for (int j = 0; j < 16; ++j) { int cc = kmapA(h, j); sc[j] = scale1[cc]; sf[j] = shift1[cc]; }
  float cb[8], s[8], q[8];
  #pragma unroll
  for (int nt = 0; nt < 8; ++nt) { cb[nt] = c1b[16*nt + nn]; s[nt] = 0; q[nt] = 0; }
  for (int tile = wave; tile < NTILE; tile += nw) {
    const float* rt = r + (size_t)tile * 512;
    BF A0, A1;                                        // z^T A-frags, K=0..31 / 32..63
    #pragma unroll
    for (int j = 0; j < 16; ++j) {
      float z = rt[kmapA(h, j) * 16 + nn] * sc[j] + sf[j];
      A0.e[j] = f2bf(mishf(z));
      A1.e[j] = f2bf(mishf(-z));
    }
    #pragma unroll
    for (int nt = 0; nt < 8; ++nt) {
      v8f D = {0,0,0,0,0,0,0,0};
      D = wmma_bf16(A0.b, load_frag(wf, FRAG_WC1 + nt,     lane).b, D);
      D = wmma_bf16(A1.b, load_frag(wf, FRAG_WC1 + 8 + nt, lane).b, D);
      float* up = u + (size_t)tile * 2048 + (16*nt + nn) * 16 + 8*h;
      float vs[8];
      #pragma unroll
      for (int v = 0; v < 8; ++v) vs[v] = D[v] + cb[nt];
      f4 o0 = {vs[0],vs[1],vs[2],vs[3]}, o1 = {vs[4],vs[5],vs[6],vs[7]};
      *(f4*)up = o0; *(f4*)(up + 4) = o1;
      #pragma unroll
      for (int v = 0; v < 8; ++v) { s[nt] += vs[v]; q[nt] += vs[v]*vs[v]; }
    }
  }
  #pragma unroll
  for (int nt = 0; nt < 8; ++nt) {
    sh[(wid*32 + lane)*16 + nt*2 + 0] = s[nt];
    sh[(wid*32 + lane)*16 + nt*2 + 1] = q[nt];
  }
  __syncthreads();
  {
    int c = threadIdx.x >> 1, k = threadIdx.x & 1;    // c in [0,128)
    int nt = c >> 4, cl = c & 15;
    float acc = 0;
    for (int w = 0; w < 8; ++w)
      for (int p = 0; p < 2; ++p)
        acc += sh[(w*32 + cl + 16*p)*16 + nt*2 + k];
    pbuf[blockIdx.x * 256 + c*2 + k] = acc;
  }
}

//==================== end stage 2: out = Wc2 @ concat_mish(bn2(u)) + b ====================
__global__ void end2_k(const float* __restrict__ u, float* __restrict__ out,
                       const __bf16* __restrict__ wf,
                       const float* __restrict__ c2b,
                       const float* __restrict__ scale2, const float* __restrict__ shift2) {
  __shared__ float ssc[128], ssf[128];
  if (threadIdx.x < 128) { ssc[threadIdx.x] = scale2[threadIdx.x]; ssf[threadIdx.x] = shift2[threadIdx.x]; }
  __syncthreads();
  int lane = threadIdx.x & 31;
  int wave = (blockIdx.x * blockDim.x + threadIdx.x) >> 5;
  int nw = (gridDim.x * blockDim.x) >> 5;
  int h = lane >> 4, nn = lane & 15;
  for (int tile = wave; tile < NTILE; tile += nw) {
    const float* ut = u + (size_t)tile * 2048;
    BF A[8];                                          // v^T A-frags (16 pos x 256 ch)
    #pragma unroll
    for (int kt = 0; kt < 4; ++kt) {
      #pragma unroll
      for (int j = 0; j < 16; ++j) {
        int cc = 32*kt + kmapA(h, j);
        float z = ut[cc * 16 + nn] * ssc[cc] + ssf[cc];
        A[kt].e[j]     = f2bf(mishf(z));
        A[kt + 4].e[j] = f2bf(mishf(-z));
      }
    }
    int n0 = tile * 16, bidx = n0 >> 13, l0 = n0 & 8191;
    for (int nt = 0; nt < 16; ++nt) {
      v8f D = {0,0,0,0,0,0,0,0};
      #pragma unroll
      for (int kt = 0; kt < 8; ++kt)
        D = wmma_bf16(A[kt].b, load_frag(wf, FRAG_WC2 + kt*16 + nt, lane).b, D);
      int c = 16*nt + nn;
      float bias = c2b[c];
      float* op = out + (size_t)bidx * 2097152 + (size_t)c * 8192 + l0 + 8*h;
      f4 o0 = {D[0]+bias, D[1]+bias, D[2]+bias, D[3]+bias};
      f4 o1 = {D[4]+bias, D[5]+bias, D[6]+bias, D[7]+bias};
      *(f4*)op = o0; *(f4*)(op + 4) = o1;
    }
  }
}

extern "C" void kernel_launch(void* const* d_in, const int* in_sizes, int n_in,
                              void* d_out, int out_size, void* d_ws, size_t ws_size,
                              hipStream_t stream) {
  (void)in_sizes; (void)n_in; (void)out_size; (void)ws_size;
  const float* x   = (const float*)d_in[0];
  const float* sw  = (const float*)d_in[1];
  const float* sb  = (const float*)d_in[2];
  const float* gw  = (const float*)d_in[3];
  const float* gb  = (const float*)d_in[4];
  const float* rw  = (const float*)d_in[5];
  const float* rb  = (const float*)d_in[6];
  const float* bng = (const float*)d_in[7];
  const float* bnb = (const float*)d_in[8];
  const float* e1g = (const float*)d_in[9];
  const float* e1b = (const float*)d_in[10];
  const float* c1w = (const float*)d_in[11];
  const float* c1b = (const float*)d_in[12];
  const float* e2g = (const float*)d_in[13];
  const float* e2b = (const float*)d_in[14];
  const float* c2w = (const float*)d_in[15];
  const float* c2b = (const float*)d_in[16];
  float* out = (float*)d_out;

  char* ws = (char*)d_ws;
  float*   r     = (float*)(ws);                                   //  8 MB
  float*   t     = (float*)(ws + ((size_t)8  << 20));              //  8 MB
  float*   u     = (float*)(ws + ((size_t)16 << 20));              // 32 MB
  __bf16*  wf    = (__bf16*)(ws + ((size_t)48 << 20));             // 280*512*2 B
  float*   pbuf  = (float*)(ws + ((size_t)48 << 20) + (1u << 19)); // 256*256*4 B
  float*   scale = (float*)(ws + ((size_t)49 << 20));
  float*   shift = scale + 128;

  prep_frags_k<<<(NFRAG * 32 + 255) / 256, 256, 0, stream>>>(sw, gw, rw, c1w, c2w, wf);
  start_k<<<256, 256, 0, stream>>>(x, wf, sb, r);

  for (int i = 0; i < 40; ++i) {
    layer_k<<<128, 256, 0, stream>>>(r, t, wf, gb, rb, i, pbuf);
    finalize_bn_k<<<1, 128, 0, stream>>>(pbuf, 128, 32, bng + i*32, bnb + i*32, scale, shift);
    bnres_k<<<256, 256, 0, stream>>>(r, t, scale, shift);
  }

  stats_r_k<<<128, 256, 0, stream>>>(r, pbuf);
  finalize_bn_k<<<1, 128, 0, stream>>>(pbuf, 128, 32, e1g, e1b, scale, shift);
  end1_k<<<256, 256, 0, stream>>>(r, u, wf, c1b, scale, shift, pbuf);
  finalize_bn_k<<<1, 128, 0, stream>>>(pbuf, 256, 128, e2g, e2b, scale, shift);
  end2_k<<<256, 256, 0, stream>>>(u, out, wf, c2b, scale, shift);
}
